// NeighborhoodAttention1D_diffusion_20761871909015
// MI455X (gfx1250) — matrix operand order repaired and checked by reading, not compiled, not run on hardware
//
#include <hip/hip_runtime.h>
#include <hip/hip_bf16.h>

#define H_SP 313
#define W_SP 88
#define NA_K 7
#define HEADS 8
#define DIM 256
#define HEAD_DIM 32
#define M_TOTAL (H_SP * W_SP)   /* 27544 */
#define M_PAD 27552             /* multiple of 32 */
#define LN_EPS 1e-5f

typedef __attribute__((ext_vector_type(16))) __bf16        v16bf;
typedef __attribute__((ext_vector_type(8)))  float         v8f;
typedef __attribute__((ext_vector_type(4)))  unsigned int  u32x4;

union FragBF { v16bf v; u32x4 q[2]; };

__device__ __forceinline__ unsigned short f2bf(float f) {
    unsigned u = __float_as_uint(f);
    u += 0x7fffu + ((u >> 16) & 1u);          // round-to-nearest-even
    return (unsigned short)(u >> 16);
}

// ---------------------------------------------------------------------------
// Kernel 1: time embedding  -> scale_t, shift_t  (1 block, 256 threads)
// ---------------------------------------------------------------------------
__global__ void time_embed_kernel(const int* __restrict__ t,
                                  const float* __restrict__ ln_w,
                                  const float* __restrict__ ln_b,
                                  float* __restrict__ scale,
                                  float* __restrict__ shift) {
    __shared__ float s[DIM];
    const int tid = threadIdx.x;
    const float tv = (float)t[0] * (4000.0f / 100.0f);
    if (tid < 128) {
        float fr = __expf((float)tid * (-9.210340371976184f / 127.0f));
        float e  = tv * fr;
        float sn = __sinf(e), cs = __cosf(e);
        s[tid]       = sn / (1.0f + __expf(-sn));   // silu
        s[tid + 128] = cs / (1.0f + __expf(-cs));
    }
    __syncthreads();
    #pragma unroll
    for (int rr = 0; rr < 2; ++rr) {
        int r = tid + rr * 256;
        const float* w = ln_w + (size_t)r * DIM;
        float acc = ln_b[r];
        for (int k = 0; k < DIM; ++k) acc = fmaf(s[k], w[k], acc);
        if (r < 256) scale[r] = acc; else shift[r - 256] = acc;
    }
}

// ---------------------------------------------------------------------------
// Kernel 2: pack fp32 weight (N x Kd, row-major) -> bf16 N-major: wt[n*Kd+k]
// ---------------------------------------------------------------------------
__global__ void pack_w_kernel(const float* __restrict__ w,
                              unsigned short* __restrict__ wt,
                              int N, int Kd) {
    int idx = blockIdx.x * blockDim.x + threadIdx.x;
    if (idx >= N * Kd) return;
    int n = idx / Kd, k = idx - n * Kd;
    wt[idx] = f2bf(w[(size_t)n * Kd + k]);
}

// ---------------------------------------------------------------------------
// Kernel 3: LayerNorm + diffusion modulation, fp32 -> bf16 (one row per block)
// ---------------------------------------------------------------------------
__global__ void ln_mod_kernel(const float* __restrict__ x,
                              const float* __restrict__ scale,
                              const float* __restrict__ shift,
                              unsigned short* __restrict__ xf) {
    const int row = blockIdx.x;
    const int tid = threadIdx.x;
    if (row >= M_TOTAL) { xf[(size_t)row * DIM + tid] = 0; return; }  // zero pad rows
    float v  = x[(size_t)row * DIM + tid];
    float s1 = v, s2 = v * v;
    #pragma unroll
    for (int o = 16; o > 0; o >>= 1) {
        s1 += __shfl_xor(s1, o, 32);
        s2 += __shfl_xor(s2, o, 32);
    }
    __shared__ float a1[8], a2[8], mv[2];
    int w = tid >> 5, ln = tid & 31;
    if (ln == 0) { a1[w] = s1; a2[w] = s2; }
    __syncthreads();
    if (tid == 0) {
        float t1 = 0.f, t2 = 0.f;
        for (int i = 0; i < 8; ++i) { t1 += a1[i]; t2 += a2[i]; }
        float mu = t1 * (1.0f / DIM);
        mv[0] = mu;
        mv[1] = t2 * (1.0f / DIM) - mu * mu;
    }
    __syncthreads();
    float xn = (v - mv[0]) * rsqrtf(mv[1] + LN_EPS);
    xn = xn * (1.0f + scale[tid]) + shift[tid];
    xf[(size_t)row * DIM + tid] = f2bf(xn);
}

// ---------------------------------------------------------------------------
// Kernel 4: bf16 WMMA GEMM.  C[M,N] = A[Mpad,256] * Wt(K-major)[256,N] + bias
// One wave computes a 32x64 tile: 2 A frags x 4 B frags = 8 WMMAs per k-step,
// so each B fragment is reused twice and the load:wmma ratio is 12:8.
// Fragment packing follows the CDNA5 VGPR layouts:
//   A (16x32 bf16): lane<16 -> row=lane, K = [kb..kb+7] and [kb+16..kb+23]
//                   lane>=16 -> row=lane-16, K = [kb+8..kb+15] and [kb+24..kb+31]
//   B (32x16 bf16): col N=lane%16, K same as A (table order)
//   C (16x16 f32):  vgpr r -> M = r + 8*(lane/16), N = lane%16
// qscale_cols: columns < this get * HEAD_DIM^-0.5 (q-scaling for the QKV GEMM)
// ---------------------------------------------------------------------------
__global__ void gemm_bf16_wmma(const unsigned short* __restrict__ A,
                               const unsigned short* __restrict__ Wt,
                               const float* __restrict__ bias,
                               float* __restrict__ C,
                               int M, int N, int Ngroups, int qscale_cols) {
    const int wid  = (blockIdx.x * blockDim.x + threadIdx.x) >> 5;
    const int lane = threadIdx.x & 31;
    const int mt = wid / Ngroups;
    const int ng = wid - mt * Ngroups;
    const int m0 = mt << 5;                       // 32 rows per wave
    if (m0 >= M_PAD) return;
    const int n0 = ng << 6;
    const int hf = lane >> 4;
    const int r  = lane & 15;
    const unsigned short* arow0 = A + (size_t)(m0 + r) * DIM + hf * 8;
    const unsigned short* arow1 = arow0 + (size_t)16 * DIM;
    const unsigned short* brow  = Wt + (size_t)(n0 + r) * DIM + hf * 8;

    v8f acc[8] = {};
    #pragma unroll 2
    for (int kb = 0; kb < DIM; kb += 32) {
        FragBF a0, a1;
        a0.q[0] = *(const u32x4*)(arow0 + kb);
        a0.q[1] = *(const u32x4*)(arow0 + kb + 16);
        a1.q[0] = *(const u32x4*)(arow1 + kb);
        a1.q[1] = *(const u32x4*)(arow1 + kb + 16);
        FragBF b[4];
        #pragma unroll
        for (int j = 0; j < 4; ++j) {
            b[j].q[0] = *(const u32x4*)(brow + (size_t)j * 16 * DIM + kb);
            b[j].q[1] = *(const u32x4*)(brow + (size_t)j * 16 * DIM + kb + 16);
        }
        #pragma unroll
        for (int j = 0; j < 4; ++j) {
            acc[j]     = __builtin_amdgcn_wmma_f32_16x16x32_bf16(
                false, a0.v, false, b[j].v, (short)0, acc[j],     false, false);
            acc[j + 4] = __builtin_amdgcn_wmma_f32_16x16x32_bf16(
                false, a1.v, false, b[j].v, (short)0, acc[j + 4], false, false);
        }

    }

    const float qs = 0.17677669529663687f;  // 32^-0.5
    #pragma unroll
    for (int mh = 0; mh < 2; ++mh) {
        #pragma unroll
        for (int j = 0; j < 4; ++j) {
            int col = n0 + j * 16 + r;
            float bval = bias[col];
            #pragma unroll
            for (int r8 = 0; r8 < 8; ++r8) {
                int row = m0 + mh * 16 + r8 + hf * 8;
                if (row < M) {
                    float vv = acc[mh * 4 + j][r8] + bval;
                    if (col < qscale_cols) vv *= qs;
                    C[(size_t)row * N + col] = vv;
                }
            }
        }
    }
}

// ---------------------------------------------------------------------------
// Kernel 5: neighborhood attention (K=7, HEAD_DIM=32 = one wave per (pos,head))
// ---------------------------------------------------------------------------
__global__ void natt_kernel(const float* __restrict__ qkv,
                            const float* __restrict__ rpb,
                            unsigned short* __restrict__ attn) {
    const int wid  = (blockIdx.x * blockDim.x + threadIdx.x) >> 5;
    const int lane = threadIdx.x & 31;
    const int h  = wid & 7;
    const int bl = wid >> 3;
    if (bl >= M_PAD) return;
    if (bl >= M_TOTAL) { attn[(size_t)bl * DIM + h * HEAD_DIM + lane] = 0; return; }
    const int l    = bl % H_SP;
    const int base = bl - l;  // b * L
    const float qd = qkv[(size_t)bl * 768 + h * HEAD_DIM + lane];  // pre-scaled q
    int start = l - (NA_K / 2);
    if (start < 0) start = 0;
    if (start > H_SP - NA_K) start = H_SP - NA_K;

    float sc[NA_K];
    float smax = -3.0e38f;
    #pragma unroll
    for (int j = 0; j < NA_K; ++j) {
        int nb = start + j;
        float kd = qkv[(size_t)(base + nb) * 768 + 256 + h * HEAD_DIM + lane];
        float p  = qd * kd;
        #pragma unroll
        for (int o = 16; o > 0; o >>= 1) p += __shfl_xor(p, o, 32);
        p += rpb[h * (2 * NA_K - 1) + (nb - l + NA_K - 1)];
        sc[j] = p;
        smax  = fmaxf(smax, p);
    }
    float sum = 0.f;
    #pragma unroll
    for (int j = 0; j < NA_K; ++j) { sc[j] = __expf(sc[j] - smax); sum += sc[j]; }
    const float inv = 1.0f / sum;
    float o = 0.f;
    #pragma unroll
    for (int j = 0; j < NA_K; ++j)
        o += sc[j] * qkv[(size_t)(base + start + j) * 768 + 512 + h * HEAD_DIM + lane];
    attn[(size_t)bl * DIM + h * HEAD_DIM + lane] = f2bf(o * inv);
}

// ---------------------------------------------------------------------------
// Kernel 6: copy t into the tail of the output tuple
// ---------------------------------------------------------------------------
__global__ void write_t_kernel(const int* __restrict__ t, float* __restrict__ out) {
    out[0] = (float)t[0];
}

// ---------------------------------------------------------------------------
extern "C" void kernel_launch(void* const* d_in, const int* in_sizes, int n_in,
                              void* d_out, int out_size, void* d_ws, size_t ws_size,
                              hipStream_t stream) {
    (void)in_sizes; (void)n_in; (void)ws_size;
    const float* x      = (const float*)d_in[0];
    const int*   t      = (const int*)d_in[1];
    const float* qkv_w  = (const float*)d_in[2];
    const float* qkv_b  = (const float*)d_in[3];
    const float* proj_w = (const float*)d_in[4];
    const float* proj_b = (const float*)d_in[5];
    const float* ln_w   = (const float*)d_in[6];
    const float* ln_b   = (const float*)d_in[7];
    const float* rpb    = (const float*)d_in[8];
    float* out = (float*)d_out;

    char* ws = (char*)d_ws;
    size_t off = 0;
    float* scale = (float*)(ws + off);            off += 256 * 4;
    float* shift = (float*)(ws + off);            off += 256 * 4;
    unsigned short* qkvw_t  = (unsigned short*)(ws + off); off += (size_t)768 * 256 * 2;
    unsigned short* projw_t = (unsigned short*)(ws + off); off += (size_t)256 * 256 * 2;
    unsigned short* xf      = (unsigned short*)(ws + off); off += (size_t)M_PAD * DIM * 2;
    unsigned short* attnb   = (unsigned short*)(ws + off); off += (size_t)M_PAD * DIM * 2;
    float* qkvbuf           = (float*)(ws + off);          off += (size_t)M_PAD * 768 * 4;

    time_embed_kernel<<<1, 256, 0, stream>>>(t, ln_w, ln_b, scale, shift);
    pack_w_kernel<<<(768 * 256 + 255) / 256, 256, 0, stream>>>(qkv_w, qkvw_t, 768, 256);
    pack_w_kernel<<<(256 * 256 + 255) / 256, 256, 0, stream>>>(proj_w, projw_t, 256, 256);
    ln_mod_kernel<<<M_PAD, 256, 0, stream>>>(x, scale, shift, xf);

    // QKV GEMM: 861 M-tiles (32 rows) x 12 N-groups (64 cols) = 10332 waves
    gemm_bf16_wmma<<<(861 * 12 + 7) / 8, 256, 0, stream>>>(
        xf, qkvw_t, qkv_b, qkvbuf, M_TOTAL, 768, 12, 256);

    natt_kernel<<<M_PAD, 256, 0, stream>>>(qkvbuf, rpb, attnb);

    // Projection GEMM: 861 x 4 = 3444 waves, writes y to d_out
    gemm_bf16_wmma<<<(861 * 4 + 7) / 8, 256, 0, stream>>>(
        attnb, projw_t, proj_b, out, M_TOTAL, 256, 4, 0);

    if (out_size > M_TOTAL * DIM)
        write_t_kernel<<<1, 1, 0, stream>>>(t, out + (size_t)M_TOTAL * DIM);
}
